// OIMArcSoftmax_54168127537872
// MI455X (gfx1250) — matrix operand order repaired and checked by reading, not compiled, not run on hardware
//
#include <hip/hip_runtime.h>
#include <hip/hip_bf16.h>
#include <math.h>

typedef unsigned short ushort_t;
typedef unsigned int   uint_t;

typedef __attribute__((ext_vector_type(16))) __bf16 v16bf;
typedef __attribute__((ext_vector_type(8)))  float  v8f;
typedef __attribute__((ext_vector_type(4)))  uint_t uint4v;
typedef __attribute__((ext_vector_type(4)))  float  float4v;

#define D_DIM 256
#define LDS_STRIDE 280                        /* elements; 560B rows: 16B-aligned, bank-conflict-free */
#define ARC_S       30.0f
#define ARC_COS_M   0.8775825618903728f
#define ARC_SIN_M   0.4794255386042030f
#define ARC_THRESH  (-0.8775825618903728f)   /* cos(pi - 0.5) */
#define ARC_MM      0.2397127693021015f      /* sin(pi-0.5)*0.5 */
#define NRM_EPS     1e-12f

__device__ __forceinline__ ushort_t f32_to_bf16_rn(float f) {
    uint_t u = __float_as_uint(f);
    uint_t lsb = (u >> 16) & 1u;
    u += 0x7fffu + lsb;
    return (ushort_t)(u >> 16);
}
__device__ __forceinline__ float bf16_to_f32(ushort_t h) {
    return __uint_as_float(((uint_t)h) << 16);
}

// ---------------------------------------------------------------- K1: row-normalize features, emit f32 + bf16 hi/lo split
__global__ void k_normalize(const float* __restrict__ feat,
                            float* __restrict__ f,
                            ushort_t* __restrict__ fhi,
                            ushort_t* __restrict__ flo) {
    __shared__ float red[D_DIM];
    const int row = blockIdx.x, tid = threadIdx.x;
    const float x = feat[(size_t)row * D_DIM + tid];
    red[tid] = x * x;
    __syncthreads();
    for (int s = D_DIM / 2; s > 0; s >>= 1) {
        if (tid < s) red[tid] += red[tid + s];
        __syncthreads();
    }
    const float inv = 1.0f / fmaxf(sqrtf(red[0]), NRM_EPS);
    const float y = x * inv;
    f  [(size_t)row * D_DIM + tid] = y;
    const ushort_t h = f32_to_bf16_rn(y);
    fhi[(size_t)row * D_DIM + tid] = h;
    flo[(size_t)row * D_DIM + tid] = f32_to_bf16_rn(y - bf16_to_f32(h));
}

// ---------------------------------------------------------------- K2: target logit + per-row margin terms
__global__ void k_target_logit(const float* __restrict__ f,
                               const float* __restrict__ weight,
                               const int* __restrict__ targets,
                               float* __restrict__ tl,
                               float* __restrict__ ctm,
                               float* __restrict__ ftl) {
    __shared__ float red[D_DIM];
    const int row = blockIdx.x, tid = threadIdx.x;
    const int tgt = targets[row];
    red[tid] = f[(size_t)row * D_DIM + tid] * weight[(size_t)tgt * D_DIM + tid];
    __syncthreads();
    for (int s = D_DIM / 2; s > 0; s >>= 1) {
        if (tid < s) red[tid] += red[tid + s];
        __syncthreads();
    }
    if (tid == 0) {
        const float t = red[0];
        tl[row] = t;
        const float sin_t = sqrtf(fmaxf(1.0f - t * t, 0.0f));
        const float cm = t * ARC_COS_M - sin_t * ARC_SIN_M;
        ctm[row] = cm;
        ftl[row] = (t > ARC_THRESH) ? cm : (t - ARC_MM);
    }
}

// ---------------------------------------------------------------- K3: EMA scalar t_new
__global__ void k_tnew(const float* __restrict__ tl,
                       const float* __restrict__ t_in,
                       float* __restrict__ tnew_ws,
                       float* __restrict__ out_t, int B) {
    __shared__ float red[512];
    const int tid = threadIdx.x;
    red[tid] = (tid < B) ? tl[tid] : 0.0f;
    __syncthreads();
    for (int s = 256; s > 0; s >>= 1) {
        if (tid < s) red[tid] += red[tid + s];
        __syncthreads();
    }
    if (tid == 0) {
        const float v = (red[0] / (float)B) * 0.01f + 0.99f * t_in[0];
        tnew_ws[0] = v;
        out_t[0]   = v;
    }
}

// ---------------------------------------------------------------- K4: WMMA bf16x3 GEMM (f @ W^T) + ArcFace epilogue
// One block (8 waves) per 16-column tile. The f32 W tile is converted to bf16 hi/lo
// ONCE into LDS (in B-fragment layout), then all 32 row-tiles consume it.
__global__ void __launch_bounds__(256)
k_gemm_arc(const ushort_t* __restrict__ fhi,
           const ushort_t* __restrict__ flo,
           const float* __restrict__ weight,
           const int* __restrict__ targets,
           const float* __restrict__ ctm,
           const float* __restrict__ ftl,
           const float* __restrict__ tnew,
           float* __restrict__ out, int B, int C) {
    __shared__ ushort_t whi[16 * LDS_STRIDE];
    __shared__ ushort_t wlo[16 * LDS_STRIDE];

    const int tid   = threadIdx.x;
    const int cbase = blockIdx.x * 16;

    // ---- Stage 1: load 16x256 f32 W tile, split to bf16 hi/lo in LDS (16 floats/thread)
    {
        const int r  = tid >> 4;           // output column within tile (B row-major in LDS)
        const int kc = (tid & 15) * 16;    // k chunk base
        const float* src  = weight + (size_t)(cbase + r) * D_DIM + kc;
        ushort_t*    dsth = whi + r * LDS_STRIDE + kc;
        ushort_t*    dstl = wlo + r * LDS_STRIDE + kc;
        #pragma unroll
        for (int i = 0; i < 16; i += 4) {
            const float4v q = *(const float4v*)(src + i);
            #pragma unroll
            for (int j = 0; j < 4; ++j) {
                const float    x = q[j];
                const ushort_t h = f32_to_bf16_rn(x);
                dsth[i + j] = h;
                dstl[i + j] = f32_to_bf16_rn(x - bf16_to_f32(h));
            }
        }
    }
    __syncthreads();

    // ---- Stage 2: each wave computes 16x16 tiles for row-tiles wave, wave+8, ...
    const int lane = tid & 31;
    const int wave = tid >> 5;             // 0..7
    const int half = lane >> 4;            // 0 or 1
    const int l16  = lane & 15;
    const int col  = cbase + l16;          // this lane's B/D column
    const float tn = tnew[0];

    // B fragment: lane -> column l16; per k-step, K = kb + 8*half + 0..7 and kb + 16 + 8*half + 0..7
    const ushort_t* bh_base = whi + l16 * LDS_STRIDE + 8 * half;
    const ushort_t* bl_base = wlo + l16 * LDS_STRIDE + 8 * half;

    union V16 { v16bf v; uint4v q[2]; };

    const int ntiles = B / 16;             // 32
    for (int rt = wave; rt < ntiles; rt += 8) {
        const int rbase = rt * 16;
        // A fragment (16-bit 16x32): lane -> row l16; VGPR0-3: K=8*half+0..7, VGPR4-7: K=16+8*half+0..7
        const ushort_t* arow_hi = fhi + (size_t)(rbase + l16) * D_DIM;
        const ushort_t* arow_lo = flo + (size_t)(rbase + l16) * D_DIM;

        v8f acc0 = {};   // hi*hi
        v8f acc1 = {};   // cross terms
        #pragma unroll
        for (int kb = 0; kb < D_DIM; kb += 32) {
            V16 Ah, Al, Bh, Bl;
            Ah.q[0] = *(const uint4v*)(arow_hi + kb +      8 * half);
            Ah.q[1] = *(const uint4v*)(arow_hi + kb + 16 + 8 * half);
            Al.q[0] = *(const uint4v*)(arow_lo + kb +      8 * half);
            Al.q[1] = *(const uint4v*)(arow_lo + kb + 16 + 8 * half);
            Bh.q[0] = *(const uint4v*)(bh_base + kb);
            Bh.q[1] = *(const uint4v*)(bh_base + kb + 16);
            Bl.q[0] = *(const uint4v*)(bl_base + kb);
            Bl.q[1] = *(const uint4v*)(bl_base + kb + 16);
            // split-precision: hi*hi into acc0; lo*hi + hi*lo into acc1 (independent chains)
            acc0 = __builtin_amdgcn_wmma_f32_16x16x32_bf16(false, Ah.v, false, Bh.v, (short)0, acc0, false, false);
            acc1 = __builtin_amdgcn_wmma_f32_16x16x32_bf16(false, Al.v, false, Bh.v, (short)0, acc1, false, false);
            acc1 = __builtin_amdgcn_wmma_f32_16x16x32_bf16(false, Ah.v, false, Bl.v, (short)0, acc1, false, false);
        }

        // ---- Epilogue: D layout -> VGPR v holds row M = 8*half + v, col = l16
        #pragma unroll
        for (int v = 0; v < 8; ++v) {
            const int row = rbase + 8 * half + v;
            float val = acc0[v] + acc1[v];
            if (val > ctm[row]) val = val * (tn + val);   // hard-example reweight
            if (col == targets[row]) val = ftl[row];      // scatter final target logit
            out[(size_t)row * C + col] = val * ARC_S;
        }
    }
}

// ---------------------------------------------------------------- K5: memory-bank EMA update
__global__ void k_zero_counts(int* __restrict__ counts, int C) {
    const int i = blockIdx.x * blockDim.x + threadIdx.x;
    if (i < C) counts[i] = 0;
}
__global__ void k_count(const int* __restrict__ targets, int* __restrict__ counts, int B) {
    const int i = blockIdx.x * blockDim.x + threadIdx.x;
    if (i < B) atomicAdd(&counts[targets[i]], 1);
}
__global__ void k_weight_update(const float* __restrict__ weight,
                                const float* __restrict__ f,
                                const int* __restrict__ targets,
                                const int* __restrict__ counts,
                                float* __restrict__ out_w, int B) {
    const int c = blockIdx.x, d = threadIdx.x;
    const float w = weight[(size_t)c * D_DIM + d];
    const int cnt = counts[c];               // block-uniform
    if (cnt == 0) {                          // untouched class: plain copy
        out_w[(size_t)c * D_DIM + d] = w;
        return;
    }
    __shared__ int   st[512];
    __shared__ float red[D_DIM];
    for (int i = d; i < B; i += D_DIM) st[i] = targets[i];
    __syncthreads();
    float s = 0.0f;
    for (int b = 0; b < B; ++b)
        if (st[b] == c) s += f[(size_t)b * D_DIM + d];
    const float upd = 0.5f * w + 0.5f * (s / (float)cnt);
    red[d] = upd * upd;
    __syncthreads();
    for (int sft = D_DIM / 2; sft > 0; sft >>= 1) {
        if (d < sft) red[d] += red[d + sft];
        __syncthreads();
    }
    const float inv = 1.0f / fmaxf(sqrtf(red[0]), NRM_EPS);
    out_w[(size_t)c * D_DIM + d] = upd * inv;
}

// ----------------------------------------------------------------
extern "C" void kernel_launch(void* const* d_in, const int* in_sizes, int n_in,
                              void* d_out, int out_size, void* d_ws, size_t ws_size,
                              hipStream_t stream) {
    const float* features = (const float*)d_in[0];
    const int*   targets  = (const int*)d_in[1];
    const float* weight   = (const float*)d_in[2];
    const float* t_in     = (const float*)d_in[3];

    const int B = in_sizes[1];                 // 512
    const int D = in_sizes[0] / B;             // 256
    const int C = in_sizes[2] / D;             // 100000

    float* out_logits = (float*)d_out;
    float* out_weight = out_logits + (size_t)B * C;
    float* out_t      = out_weight + (size_t)C * D_DIM;

    char* ws = (char*)d_ws;
    float*    f    = (float*)ws;    ws += (size_t)B * D_DIM * sizeof(float);
    ushort_t* fhi  = (ushort_t*)ws; ws += (size_t)B * D_DIM * sizeof(ushort_t);
    ushort_t* flo  = (ushort_t*)ws; ws += (size_t)B * D_DIM * sizeof(ushort_t);
    float*    tl   = (float*)ws;    ws += (size_t)B * sizeof(float);
    float*    ctm  = (float*)ws;    ws += (size_t)B * sizeof(float);
    float*    ftl  = (float*)ws;    ws += (size_t)B * sizeof(float);
    float*    tnew = (float*)ws;    ws += sizeof(float);
    int*      counts = (int*)ws;

    k_normalize   <<<B, D_DIM, 0, stream>>>(features, f, fhi, flo);
    k_target_logit<<<B, D_DIM, 0, stream>>>(f, weight, targets, tl, ctm, ftl);
    k_tnew        <<<1, 512, 0, stream>>>(tl, t_in, tnew, out_t, B);

    k_gemm_arc<<<C / 16, 256, 0, stream>>>(fhi, flo, weight, targets, ctm, ftl, tnew,
                                           out_logits, B, C);

    k_zero_counts  <<<(C + 255) / 256, 256, 0, stream>>>(counts, C);
    k_count        <<<(B + 255) / 256, 256, 0, stream>>>(targets, counts, B);
    k_weight_update<<<C, D_DIM, 0, stream>>>(weight, f, targets, counts, out_weight, B);
}
